// Attention_53798760350139
// MI455X (gfx1250) — compile-verified, matrix operand order checked by reading, not run on hardware
//
#include <hip/hip_runtime.h>
#include <stdint.h>

// ---------------------------------------------------------------------------
// Problem constants (B,H,W,D) = (16,48,48,384); N = H*W = 2304 tokens/batch.
// ---------------------------------------------------------------------------
#define DD    384
#define NTOK  2304
#define NBAT  16
#define CNT   (NTOK * DD)          // 884736 elements per batch
#define EPSLN 1e-5f

typedef __attribute__((ext_vector_type(16))) __bf16          v16bf;
typedef __attribute__((ext_vector_type(8)))  float           v8f;
typedef __attribute__((ext_vector_type(8)))  unsigned short  ushort8;

__device__ __forceinline__ unsigned short f2bf(float f) {
  union { float f; unsigned u; } c; c.f = f;
  unsigned r = c.u + 0x7fffu + ((c.u >> 16) & 1u);   // round-to-nearest-even
  return (unsigned short)(r >> 16);
}

// WMMA bf16 fragment: 16 bf16 per lane. Per the CDNA5 ISA VGPR layout,
// lane L (row/col = L&15, half = L>=16) holds K = half*8 + [0..7] in VGPRs 0-3
// and K = 16 + half*8 + [0..7] in VGPRs 4-7 -> two contiguous 16B LDS loads.
struct Frag { union { v16bf v; struct { ushort8 lo, hi; } h; }; };

__device__ __forceinline__ Frag ld_frag(const unsigned short* p, int half) {
  Frag f;
  f.h.lo = *(const ushort8*)(p + half * 8);
  f.h.hi = *(const ushort8*)(p + half * 8 + 16);
  return f;
}

// ---------------------------------------------------------------------------
// Kernel 1: x += positional encoding; emit fp32 (residual) and bf16 (GEMM).
// ---------------------------------------------------------------------------
__global__ void pos_add_k(const float* __restrict__ x,
                          const float* __restrict__ Wp,
                          const float* __restrict__ bp,
                          float* __restrict__ xf,
                          unsigned short* __restrict__ xfb) {
  long i = (long)blockIdx.x * 256 + threadIdx.x;   // < NBAT*NTOK*DD
  int d = (int)(i % DD);
  long nd = i / DD;
  int n = (int)(nd % NTOK);
  float ph = (float)(n / 48);
  float pw = (float)(n % 48);
  float v = x[i] + ph * Wp[d] + pw * Wp[DD + d] + bp[d];
  xf[i] = v;
  xfb[i] = f2bf(v);
}

// Kernel 1b: W1/W2 fp32 -> transposed bf16 (tiny, 0.3 MB each).
__global__ void wconv_k(const float* __restrict__ W1, const float* __restrict__ W2,
                        unsigned short* __restrict__ W1T, unsigned short* __restrict__ W2T) {
  int i = blockIdx.x * 256 + threadIdx.x;          // i = k*DD + n, i < DD*DD
  int k = i / DD, n = i % DD;
  W1T[n * DD + k] = f2bf(W1[i]);
  W2T[n * DD + k] = f2bf(W2[i]);
}

// Kernel 1c: tiled bf16 transpose X[b][n][d] -> XT[b][d][n] (for O = P*X).
__global__ void transpose_k(const unsigned short* __restrict__ in,
                            unsigned short* __restrict__ outT) {
  __shared__ unsigned short t[32][33];
  const int b  = blockIdx.z;
  const int n0 = blockIdx.x * 32;
  const int d0 = blockIdx.y * 32;
  const unsigned short* src = in + (size_t)b * CNT;
  unsigned short* dst = outT + (size_t)b * CNT;
  const int tx = threadIdx.x & 31, ty = threadIdx.x >> 5;   // 256 thr: ty 0..7
#pragma unroll
  for (int j = ty; j < 32; j += 8)
    t[j][tx] = src[(size_t)(n0 + j) * DD + d0 + tx];
  __syncthreads();
#pragma unroll
  for (int j = ty; j < 32; j += 8)
    dst[(size_t)(d0 + j) * NTOK + n0 + tx] = t[tx][j];
}

// ---------------------------------------------------------------------------
// 128x128-tile bf16 WMMA GEMM, 256 threads (8 wave32), BK = 32.
// Double-buffered LDS with register-staged global loads + global_prefetch.
// A is [M][K] row-major; B operand is pre-transposed: [N][K] row-major.
//  MODE 0: C fp32 store                     (scores S = X * X^T)
//  MODE 1: C bf16 store                     (O = P * X)
//  MODE 2: bf16( relu(C + bias) )           (MLP layer 1)
//  MODE 3: fp32( C + bias + resid ) + per-tile sum/sumsq partials (MLP layer 2)
// ---------------------------------------------------------------------------
template <int MODE>
__global__ __launch_bounds__(256) void gemm128_k(
    const unsigned short* __restrict__ A, long lda,
    const unsigned short* __restrict__ B, long ldb,
    void* __restrict__ Cout, long ldc,
    const float* __restrict__ bias,
    const float* __restrict__ resid,
    float* __restrict__ partials,
    int K) {
  __shared__ __align__(16) unsigned short sA[2][128 * 40];  // 40-pitch: 16B-aligned
  __shared__ __align__(16) unsigned short sB[2][128 * 40];  // rows, spread banks

  const int tid  = threadIdx.x;
  const int lane = tid & 31;
  const int w    = tid >> 5;          // 8 waves
  const int lm   = lane & 15;
  const int half = lane >> 4;
  const int wm   = w >> 1;            // 4 m-groups (2 mtiles each)
  const int wn   = w & 1;             // 2 n-groups (4 ntiles each)
  const long bM  = (long)blockIdx.y * 128;
  const long bN  = (long)blockIdx.x * 128;

  // Staging coordinates: thread covers (r0, c0) and (r0+64, c0) of the
  // 128x32 tile with 16B vector ops.
  const int r0   = tid >> 2;
  const int c0   = (tid & 3) << 3;
  const int dst0 = r0 * 40 + c0;
  const int dst1 = (r0 + 64) * 40 + c0;
  const unsigned short* pA0 = A + (bM + r0) * lda + c0;
  const unsigned short* pA1 = A + (bM + r0 + 64) * lda + c0;
  const unsigned short* pB0 = B + (bN + r0) * ldb + c0;
  const unsigned short* pB1 = B + (bN + r0 + 64) * ldb + c0;

  const v8f vz = {0.f, 0.f, 0.f, 0.f, 0.f, 0.f, 0.f, 0.f};
  v8f acc[2][4];
#pragma unroll
  for (int mt = 0; mt < 2; ++mt)
#pragma unroll
    for (int nt = 0; nt < 4; ++nt) acc[mt][nt] = vz;

  // Prologue: stage tile 0.
  {
    ushort8 a0 = *(const ushort8*)pA0;
    ushort8 a1 = *(const ushort8*)pA1;
    ushort8 b0 = *(const ushort8*)pB0;
    ushort8 b1 = *(const ushort8*)pB1;
    *(ushort8*)&sA[0][dst0] = a0;
    *(ushort8*)&sA[0][dst1] = a1;
    *(ushort8*)&sB[0][dst0] = b0;
    *(ushort8*)&sB[0][dst1] = b1;
  }
  __syncthreads();

  const int nk = K >> 5;
  for (int t = 0; t < nk; ++t) {
    const int cur = t & 1, nxt = cur ^ 1;
    const bool more = (t + 1) < nk;

    // Issue next tile's global loads early (latency overlapped with WMMA).
    ushort8 na0, na1, nb0, nb1;
    if (more) {
      const long ko = (long)(t + 1) << 5;
      na0 = *(const ushort8*)(pA0 + ko);
      na1 = *(const ushort8*)(pA1 + ko);
      nb0 = *(const ushort8*)(pB0 + ko);
      nb1 = *(const ushort8*)(pB1 + ko);
      if ((t + 2) < nk) {                 // gfx1250 global_prefetch_b8 two ahead
        __builtin_prefetch(pA0 + ko + 32, 0, 0);
        __builtin_prefetch(pB0 + ko + 32, 0, 0);
      }
    }

    // Compute on current buffer.
    Frag fa0 = ld_frag(&sA[cur][(wm * 32 + lm) * 40], half);
    Frag fa1 = ld_frag(&sA[cur][(wm * 32 + 16 + lm) * 40], half);
#pragma unroll
    for (int nt = 0; nt < 4; ++nt) {
      Frag fb = ld_frag(&sB[cur][((wn * 4 + nt) * 16 + lm) * 40], half);
      acc[0][nt] = __builtin_amdgcn_wmma_f32_16x16x32_bf16(
          false, fa0.v, false, fb.v, (short)0, acc[0][nt], false, false);
      acc[1][nt] = __builtin_amdgcn_wmma_f32_16x16x32_bf16(
          false, fa1.v, false, fb.v, (short)0, acc[1][nt], false, false);
    }

    // Store staged registers to the other buffer.
    if (more) {
      *(ushort8*)&sA[nxt][dst0] = na0;
      *(ushort8*)&sA[nxt][dst1] = na1;
      *(ushort8*)&sB[nxt][dst0] = nb0;
      *(ushort8*)&sB[nxt][dst1] = nb1;
    }
    __syncthreads();
  }

  // --- epilogue ---
  float lsum = 0.f, lsq = 0.f;
#pragma unroll
  for (int mt = 0; mt < 2; ++mt) {
#pragma unroll
    for (int nt = 0; nt < 4; ++nt) {
      long colg = bN + (wn * 4 + nt) * 16 + lm;
      float bv = (MODE >= 2) ? bias[colg] : 0.f;
#pragma unroll
      for (int v = 0; v < 8; ++v) {
        long rowg = bM + (wm * 2 + mt) * 16 + v + half * 8;
        float c = acc[mt][nt][v];
        if (MODE == 0) {
          ((float*)Cout)[rowg * ldc + colg] = c;
        } else if (MODE == 1) {
          ((unsigned short*)Cout)[rowg * ldc + colg] = f2bf(c);
        } else if (MODE == 2) {
          c = fmaxf(c + bv, 0.f);
          ((unsigned short*)Cout)[rowg * ldc + colg] = f2bf(c);
        } else {
          c += bv + resid[rowg * ldc + colg];
          ((float*)Cout)[rowg * ldc + colg] = c;
          lsum += c;
          lsq += c * c;
        }
      }
    }
  }
  if (MODE == 3) {
    __syncthreads();                       // done with sA/sB; reuse as scratch
    float* r = (float*)&sA[0][0];
    r[tid] = lsum;
    r[256 + tid] = lsq;
    __syncthreads();
    for (int o = 128; o > 0; o >>= 1) {
      if (tid < o) {
        r[tid] += r[tid + o];
        r[256 + tid] += r[256 + tid + o];
      }
      __syncthreads();
    }
    if (tid == 0) {
      long pi = (long)blockIdx.y * gridDim.x + blockIdx.x;
      partials[2 * pi] = r[0];
      partials[2 * pi + 1] = r[256];
    }
  }
}

// ---------------------------------------------------------------------------
// Row softmax: one 256-thread block per row of 2304 fp32 scores -> bf16 P.
// ---------------------------------------------------------------------------
__global__ void softmax_k(const float* __restrict__ S,
                          unsigned short* __restrict__ P) {
  const int tid = threadIdx.x;
  const long row = blockIdx.x;
  const float* s = S + row * (long)NTOK;
  __shared__ float red[256];

  float v[9];
  float m = -3.4e38f;
#pragma unroll
  for (int i = 0; i < 9; ++i) {
    v[i] = s[tid + i * 256];
    m = fmaxf(m, v[i]);
  }
  red[tid] = m;
  __syncthreads();
  for (int o = 128; o > 0; o >>= 1) {
    if (tid < o) red[tid] = fmaxf(red[tid], red[tid + o]);
    __syncthreads();
  }
  m = red[0];
  __syncthreads();

  float sum = 0.f;
#pragma unroll
  for (int i = 0; i < 9; ++i) {
    v[i] = __expf(v[i] - m);
    sum += v[i];
  }
  red[tid] = sum;
  __syncthreads();
  for (int o = 128; o > 0; o >>= 1) {
    if (tid < o) red[tid] += red[tid + o];
    __syncthreads();
  }
  float inv = 1.0f / red[0];

  unsigned short* p = P + row * (long)NTOK;
#pragma unroll
  for (int i = 0; i < 9; ++i) p[tid + i * 256] = f2bf(v[i] * inv);
}

// ---------------------------------------------------------------------------
// Per-batch mean / rstd from the MLP2 tile partials (54 tiles per batch).
// ---------------------------------------------------------------------------
__global__ void stats_k(const float* __restrict__ partials, float* __restrict__ st) {
  const int b = blockIdx.x;              // 16 blocks x 64 threads
  const int tid = threadIdx.x;
  __shared__ float rs[64], rq[64];
  float s = 0.f, q = 0.f;
  for (int i = tid; i < 54; i += 64) {
    s += partials[2 * (b * 54 + i)];
    q += partials[2 * (b * 54 + i) + 1];
  }
  rs[tid] = s; rq[tid] = q;
  __syncthreads();
  for (int o = 32; o > 0; o >>= 1) {
    if (tid < o) { rs[tid] += rs[tid + o]; rq[tid] += rq[tid + o]; }
    __syncthreads();
  }
  if (tid == 0) {
    float cnt = (float)CNT;
    float mean = rs[0] / cnt;
    float var = rq[0] / cnt - mean * mean;
    st[2 * b] = mean;
    st[2 * b + 1] = rsqrtf(var + EPSLN);
  }
}

__global__ void norm_k(float* __restrict__ out, const float* __restrict__ st) {
  long i = (long)blockIdx.x * 256 + threadIdx.x;
  int b = (int)(i / CNT);
  out[i] = (out[i] - st[2 * b]) * st[2 * b + 1];
}

// ---------------------------------------------------------------------------
// Host launch. Workspace requirement ~203 MB:
//   xf fp32 56.6MB + xf bf16 28.3MB + xfT bf16 28.3MB + attn bf16 28.3MB
//   + h1 bf16 28.3MB + S fp32 21.2MB + P bf16 10.6MB + W1T/W2T 0.6MB + tiny.
// ---------------------------------------------------------------------------
extern "C" void kernel_launch(void* const* d_in, const int* in_sizes, int n_in,
                              void* d_out, int out_size, void* d_ws, size_t ws_size,
                              hipStream_t stream) {
  const float* x  = (const float*)d_in[0];
  const float* Wp = (const float*)d_in[1];
  const float* bp = (const float*)d_in[2];
  const float* W1 = (const float*)d_in[3];
  const float* b1 = (const float*)d_in[4];
  const float* W2 = (const float*)d_in[5];
  const float* b2 = (const float*)d_in[6];
  float* out = (float*)d_out;

  char* p = (char*)d_ws;
  auto carve = [&](size_t bytes) {
    void* r = (void*)p;
    p += (bytes + 255) & ~(size_t)255;
    return r;
  };
  const size_t ALL = (size_t)NBAT * CNT;
  float*          xf    = (float*)carve(ALL * 4);
  unsigned short* xfb   = (unsigned short*)carve(ALL * 2);
  unsigned short* xfT   = (unsigned short*)carve(ALL * 2);
  unsigned short* attnb = (unsigned short*)carve(ALL * 2);
  unsigned short* h1b   = (unsigned short*)carve(ALL * 2);
  float*          Sbuf  = (float*)carve((size_t)NTOK * NTOK * 4);
  unsigned short* Pbuf  = (unsigned short*)carve((size_t)NTOK * NTOK * 2);
  unsigned short* W1T   = (unsigned short*)carve((size_t)DD * DD * 2);
  unsigned short* W2T   = (unsigned short*)carve((size_t)DD * DD * 2);
  float*          parts = (float*)carve((size_t)288 * 3 * 2 * 4);
  float*          st    = (float*)carve((size_t)NBAT * 2 * 4);

  const int elemBlocks = (int)(ALL / 256);  // 55296

  pos_add_k<<<elemBlocks, 256, 0, stream>>>(x, Wp, bp, xf, xfb);
  wconv_k<<<(DD * DD) / 256, 256, 0, stream>>>(W1, W2, W1T, W2T);
  transpose_k<<<dim3(NTOK / 32, DD / 32, NBAT), 256, 0, stream>>>(xfb, xfT);

  // Attention, one batch at a time (S/P buffers stay resident in L2).
  for (int b = 0; b < NBAT; ++b) {
    const unsigned short* Xb  = xfb + (size_t)b * CNT;
    const unsigned short* XbT = xfT + (size_t)b * CNT;
    // S = Xb * Xb^T   (M=N=2304, K=384); B operand = Xb ([N][K] row-major)
    gemm128_k<0><<<dim3(18, 18), 256, 0, stream>>>(
        Xb, DD, Xb, DD, (void*)Sbuf, NTOK, nullptr, nullptr, nullptr, DD);
    // softmax rows -> bf16 P
    softmax_k<<<NTOK, 256, 0, stream>>>(Sbuf, Pbuf);
    // O = P * Xb      (M=2304, N=384, K=2304); B operand = Xb^T
    gemm128_k<1><<<dim3(3, 18), 256, 0, stream>>>(
        Pbuf, NTOK, XbT, NTOK, (void*)(attnb + (size_t)b * CNT), DD,
        nullptr, nullptr, nullptr, NTOK);
  }

  // MLP layer 1: h1 = relu(O * W1 + b1)    (M=36864, N=K=384); B = W1^T
  gemm128_k<2><<<dim3(3, 288), 256, 0, stream>>>(
      attnb, DD, W1T, DD, (void*)h1b, DD, b1, nullptr, nullptr, DD);
  // MLP layer 2 + residual + LN partials: out = h1 * W2 + b2 + xf; B = W2^T
  gemm128_k<3><<<dim3(3, 288), 256, 0, stream>>>(
      h1b, DD, W2T, DD, (void*)out, DD, b2, xf, parts, DD);

  stats_k<<<NBAT, 64, 0, stream>>>(parts, st);
  norm_k<<<elemBlocks, 256, 0, stream>>>(out, st);
}
